// QoSGAE_77360950935944
// MI455X (gfx1250) — compile-verified, matrix operand order checked by reading, not run on hardware
//
#include <hip/hip_runtime.h>

typedef __bf16 bf16;
typedef __attribute__((ext_vector_type(16))) bf16 v16bf;
typedef __attribute__((ext_vector_type(8))) float v8f;

union BTile { v16bf v; uint4 q[2]; };

#define NN 10000
#define NE 320000
#define NP 1000000

#define DEC_THREADS 256                 // 8 waves per block
#define DEC_WAVES (DEC_THREADS / 32)
#define WD1_BYTES (256 * 256 * 2)       // 128 KB packed Wd1 in LDS

// ---------------- elementwise helpers ----------------
__global__ void k_zero(float* __restrict__ p, int n) {
  int i = blockIdx.x * 256 + threadIdx.x;
  if (i < n) p[i] = 0.f;
}

__global__ void k_cast(const float* __restrict__ in, bf16* __restrict__ out, int n) {
  int i = blockIdx.x * 256 + threadIdx.x;
  if (i < n) out[i] = (bf16)in[i];
}

// Pack row-major f32 weight W[K][N] into WMMA-B tiled bf16 layout:
// out[((nt*(K/32)+kc)*32 + lane)*16 + j]
//   lane<16 -> K-half {0..7,16..23}, lane>=16 -> {8..15,24..31}; n = nt*16 + (lane&15)
__global__ void k_pack_b(const float* __restrict__ W, bf16* __restrict__ out, int K, int N) {
  int e = blockIdx.x * 256 + threadIdx.x;
  if (e >= K * N) return;
  int j = e & 15, lane = (e >> 4) & 31, tile = e >> 9;
  int kcN = K >> 5;
  int kc = tile % kcN, nt = tile / kcN;
  int n = nt * 16 + (lane & 15);
  int k = kc * 32 + ((lane >> 4) << 3) + ((j >> 3) << 4) + (j & 7);
  out[e] = (bf16)W[k * N + n];
}

// ---------------- edge scatter (segment_sum via atomics) ----------------
// one block per edge; blockDim = C/4
__global__ void k_scatter(const float* __restrict__ h, const int* __restrict__ src,
                          const int* __restrict__ dst, const float* __restrict__ ew,
                          float* __restrict__ agg, int C) {
  int e = blockIdx.x;
  int c4 = threadIdx.x * 4;
  float w = ew[e];
  const float4 v = *(const float4*)(h + (size_t)src[e] * C + c4);
  float* o = agg + (size_t)dst[e] * C + c4;
  atomicAdd(o + 0, w * v.x);
  atomicAdd(o + 1, w * v.y);
  atomicAdd(o + 2, w * v.z);
  atomicAdd(o + 3, w * v.w);
}

// ---------------- GraphConv node update: out = op(agg@Wrel + h@Wroot + b) ----------------
// one wave per 16x16 output tile; NN = 625 * 16 exactly
template <bool RELU>
__global__ void k_node_gemm(const bf16* __restrict__ aggB, const bf16* __restrict__ hB,
                            const bf16* __restrict__ WrelP, const bf16* __restrict__ WrootP,
                            const float* __restrict__ bias,
                            float* __restrict__ outF, bf16* __restrict__ outB,
                            int Cin, int Cout) {
  const int MT = NN / 16;                                   // 625 M-tiles
  int wid = (blockIdx.x * blockDim.x + threadIdx.x) >> 5;
  int lane = threadIdx.x & 31;
  int mt = wid % MT, nt = wid / MT;
  if (nt >= (Cout >> 4)) return;

  int kcN = Cin >> 5;
  int row = mt * 16 + (lane & 15);                          // A: lanes L and L+16 hold row M=L&15
  int kb0 = (lane >> 4) << 3;                               // K sub-offset 0 or 8
  v8f acc = {};

  for (int kc = 0; kc < kcN; ++kc) {                        // agg @ W_rel
    BTile a, b;
    const bf16* ap = aggB + (size_t)row * Cin + kc * 32 + kb0;
    a.q[0] = *(const uint4*)(ap);
    a.q[1] = *(const uint4*)(ap + 16);
    const uint4* bp = (const uint4*)(WrelP + ((size_t)(nt * kcN + kc) * 32 + lane) * 16);
    b.q[0] = bp[0]; b.q[1] = bp[1];
    acc = __builtin_amdgcn_wmma_f32_16x16x32_bf16(false, a.v, false, b.v, (short)0, acc, false, false);
  }
  for (int kc = 0; kc < kcN; ++kc) {                        // + h @ W_root
    BTile a, b;
    const bf16* ap = hB + (size_t)row * Cin + kc * 32 + kb0;
    a.q[0] = *(const uint4*)(ap);
    a.q[1] = *(const uint4*)(ap + 16);
    const uint4* bp = (const uint4*)(WrootP + ((size_t)(nt * kcN + kc) * 32 + lane) * 16);
    b.q[0] = bp[0]; b.q[1] = bp[1];
    acc = __builtin_amdgcn_wmma_f32_16x16x32_bf16(false, a.v, false, b.v, (short)0, acc, false, false);
  }

  // epilogue: C/D layout -> lane holds n=nt*16+(lane&15), rows m0..m0+7
  int n = nt * 16 + (lane & 15);
  float bv = bias[n];
  int m0 = mt * 16 + kb0;
#pragma unroll
  for (int r = 0; r < 8; ++r) {
    float v = acc[r] + bv;
    if (RELU) v = fmaxf(v, 0.f);
    outF[(size_t)(m0 + r) * Cout + n] = v;
    outB[(size_t)(m0 + r) * Cout + n] = (bf16)v;
  }
}

// ---------------- fused decoder: relu(concat(z[u],z[v]) @ Wd1 + bd1) @ Wd2 + bd2 ----------------
// 8 waves/block, 16 pairs/wave. Packed Wd1 (128 KB bf16) staged once per block in LDS;
// B tiles read via ds_load_b128 with double buffering so WMMA overlaps the fetch.
__global__ __launch_bounds__(DEC_THREADS) void k_decode(
    const bf16* __restrict__ zB, const int* __restrict__ elist,
    const bf16* __restrict__ Wd1P, const float* __restrict__ bd1,
    const float* __restrict__ Wd2, const float* __restrict__ bd2p,
    float* __restrict__ out) {
  extern __shared__ uint4 sW4[];                            // 128 KB dynamic LDS

  // cooperative stage of packed Wd1: 8192 uint4, coalesced
  const uint4* gsrc = (const uint4*)Wd1P;
  for (int i = threadIdx.x; i < WD1_BYTES / 16; i += DEC_THREADS) sW4[i] = gsrc[i];
  __syncthreads();

  int wid = blockIdx.x * DEC_WAVES + (threadIdx.x >> 5);
  if (wid >= NP / 16) return;
  const bf16* sW = (const bf16*)sW4;

  int lane = threadIdx.x & 31;
  int p0 = wid * 16;
  int prow = p0 + (lane & 15);
  int u = elist[prow];
  int v = elist[NP + prow];
  int kb0 = (lane >> 4) << 3;

  // Gather A panel: 16 pairs x 256 (= z[u] ++ z[v]) into 8 K-chunks, 64 VGPRs
  BTile a[8];
#pragma unroll
  for (int kc = 0; kc < 8; ++kc) {
    int k1 = kc * 32 + kb0;
    int k2 = k1 + 16;
    const bf16* p1 = (k1 < 128) ? (zB + (size_t)u * 128 + k1) : (zB + (size_t)v * 128 + (k1 - 128));
    const bf16* p2 = (k2 < 128) ? (zB + (size_t)u * 128 + k2) : (zB + (size_t)v * 128 + (k2 - 128));
    a[kc].q[0] = *(const uint4*)p1;
    a[kc].q[1] = *(const uint4*)p2;
  }

  float partial[8];
#pragma unroll
  for (int r = 0; r < 8; ++r) partial[r] = 0.f;
  int n_lane = lane & 15;

  for (int ntile = 0; ntile < 16; ++ntile) {                // hidden = 256 -> 16 N-tiles
    v8f acc = {};
    const uint4* btile = (const uint4*)(sW + ((size_t)(ntile * 8) * 32 + lane) * 16);
    BTile bcur, bnxt;
    bcur.q[0] = btile[0];
    bcur.q[1] = btile[1];
#pragma unroll
    for (int kc = 0; kc < 8; ++kc) {
      if (kc + 1 < 8) {                                     // prefetch next B under this WMMA
        bnxt.q[0] = btile[(kc + 1) * 64 + 0];
        bnxt.q[1] = btile[(kc + 1) * 64 + 1];
      }
      acc = __builtin_amdgcn_wmma_f32_16x16x32_bf16(false, a[kc].v, false, bcur.v, (short)0, acc,
                                                    false, false);
      bcur = bnxt;
    }
    int n = ntile * 16 + n_lane;
    float b1 = bd1[n];
    float w2 = Wd2[n];
#pragma unroll
    for (int r = 0; r < 8; ++r)
      partial[r] += fmaxf(acc[r] + b1, 0.f) * w2;           // fused relu + second GEMM
  }

  // reduce over the 16 n-lanes in each half-wave
#pragma unroll
  for (int off = 8; off >= 1; off >>= 1)
#pragma unroll
    for (int r = 0; r < 8; ++r)
      partial[r] += __shfl_xor(partial[r], off, 32);

  if ((lane & 15) == 0) {
    float b2 = bd2p[0];
    int mbase = p0 + kb0;                                    // lane0 -> rows 0..7, lane16 -> rows 8..15
#pragma unroll
    for (int r = 0; r < 8; ++r) out[mbase + r] = partial[r] + b2;
  }
}

// ---------------- launcher ----------------
extern "C" void kernel_launch(void* const* d_in, const int* in_sizes, int n_in,
                              void* d_out, int out_size, void* d_ws, size_t ws_size,
                              hipStream_t stream) {
  const float* x     = (const float*)d_in[0];
  const int*   eidx  = (const int*)d_in[1];
  const float* ew    = (const float*)d_in[2];
  const int*   elist = (const int*)d_in[3];
  const float* Wrel1 = (const float*)d_in[4];
  const float* brel1 = (const float*)d_in[5];
  const float* Wroot1= (const float*)d_in[6];
  const float* Wrel2 = (const float*)d_in[7];
  const float* brel2 = (const float*)d_in[8];
  const float* Wroot2= (const float*)d_in[9];
  const float* Wrel3 = (const float*)d_in[10];
  const float* brel3 = (const float*)d_in[11];
  const float* Wroot3= (const float*)d_in[12];
  const float* Wd1   = (const float*)d_in[13];
  const float* bd1   = (const float*)d_in[14];
  const float* Wd2   = (const float*)d_in[15];
  const float* bd2   = (const float*)d_in[16];
  float* out = (float*)d_out;

  const int* src = eidx;
  const int* dst = eidx + NE;

  char* w = (char*)d_ws;
  auto alloc = [&](size_t bytes) -> void* {
    void* p = (void*)w;
    w += (bytes + 255) & ~(size_t)255;
    return p;
  };

  bf16*  xB      = (bf16*)alloc((size_t)NN * 256 * 2);
  float* h1F     = (float*)alloc((size_t)NN * 256 * 4);
  bf16*  h1B     = (bf16*)alloc((size_t)NN * 256 * 2);
  float* h2F     = (float*)alloc((size_t)NN * 128 * 4);
  bf16*  h2B     = (bf16*)alloc((size_t)NN * 128 * 2);
  float* zF      = (float*)alloc((size_t)NN * 128 * 4);
  bf16*  zB      = (bf16*)alloc((size_t)NN * 128 * 2);
  float* agg     = (float*)alloc((size_t)NN * 256 * 4);
  bf16*  aggB    = (bf16*)alloc((size_t)NN * 256 * 2);
  bf16*  Wrel1P  = (bf16*)alloc(256 * 256 * 2);
  bf16*  Wroot1P = (bf16*)alloc(256 * 256 * 2);
  bf16*  Wrel2P  = (bf16*)alloc(256 * 128 * 2);
  bf16*  Wroot2P = (bf16*)alloc(256 * 128 * 2);
  bf16*  Wrel3P  = (bf16*)alloc(128 * 128 * 2);
  bf16*  Wroot3P = (bf16*)alloc(128 * 128 * 2);
  bf16*  Wd1P    = (bf16*)alloc(256 * 256 * 2);

  auto grid = [](int n) { return (n + 255) / 256; };

  // weight packing + input cast
  k_cast<<<grid(NN * 256), 256, 0, stream>>>(x, xB, NN * 256);
  k_pack_b<<<grid(256 * 256), 256, 0, stream>>>(Wrel1, Wrel1P, 256, 256);
  k_pack_b<<<grid(256 * 256), 256, 0, stream>>>(Wroot1, Wroot1P, 256, 256);
  k_pack_b<<<grid(256 * 128), 256, 0, stream>>>(Wrel2, Wrel2P, 256, 128);
  k_pack_b<<<grid(256 * 128), 256, 0, stream>>>(Wroot2, Wroot2P, 256, 128);
  k_pack_b<<<grid(128 * 128), 256, 0, stream>>>(Wrel3, Wrel3P, 128, 128);
  k_pack_b<<<grid(128 * 128), 256, 0, stream>>>(Wroot3, Wroot3P, 128, 128);
  k_pack_b<<<grid(256 * 256), 256, 0, stream>>>(Wd1, Wd1P, 256, 256);

  // ---- layer 1: 256 -> 256, relu ----
  k_zero<<<grid(NN * 256), 256, 0, stream>>>(agg, NN * 256);
  k_scatter<<<NE, 256 / 4, 0, stream>>>(x, src, dst, ew, agg, 256);
  k_cast<<<grid(NN * 256), 256, 0, stream>>>(agg, aggB, NN * 256);
  {
    int waves = (NN / 16) * (256 / 16);                     // 10000
    k_node_gemm<true><<<waves * 32 / 128, 128, 0, stream>>>(aggB, xB, Wrel1P, Wroot1P, brel1,
                                                            h1F, h1B, 256, 256);
  }

  // ---- layer 2: 256 -> 128, relu ----
  k_zero<<<grid(NN * 256), 256, 0, stream>>>(agg, NN * 256);
  k_scatter<<<NE, 256 / 4, 0, stream>>>(h1F, src, dst, ew, agg, 256);
  k_cast<<<grid(NN * 256), 256, 0, stream>>>(agg, aggB, NN * 256);
  {
    int waves = (NN / 16) * (128 / 16);                     // 5000
    k_node_gemm<true><<<waves * 32 / 128, 128, 0, stream>>>(aggB, h1B, Wrel2P, Wroot2P, brel2,
                                                            h2F, h2B, 256, 128);
  }

  // ---- layer 3: 128 -> 128, linear ----
  k_zero<<<grid(NN * 128), 256, 0, stream>>>(agg, NN * 128);
  k_scatter<<<NE, 128 / 4, 0, stream>>>(h2F, src, dst, ew, agg, 128);
  k_cast<<<grid(NN * 128), 256, 0, stream>>>(agg, aggB, NN * 128);
  {
    int waves = (NN / 16) * (128 / 16);                     // 5000
    k_node_gemm<false><<<waves * 32 / 128, 128, 0, stream>>>(aggB, h2B, Wrel3P, Wroot3P, brel3,
                                                             zF, zB, 128, 128);
  }

  // ---- fused decoder over 1M pairs: 128 pairs/block, Wd1 in LDS ----
  {
    int totalWaves = NP / 16;                               // 62500
    int blocks = (totalWaves + DEC_WAVES - 1) / DEC_WAVES;  // 7813
    k_decode<<<blocks, DEC_THREADS, WD1_BYTES, stream>>>(zB, elist, Wd1P, bd1, Wd2, bd2, out);
  }
}
